// EquivariantMLPBlock_79096117723774
// MI455X (gfx1250) — matrix operand order repaired and checked by reading, not compiled, not run on hardware
//
#include <hip/hip_runtime.h>
#include <cstdint>

typedef _Float16 v16h __attribute__((ext_vector_type(16)));
typedef _Float16 h8   __attribute__((ext_vector_type(8)));
typedef float    v8f  __attribute__((ext_vector_type(8)));

#define D_IN   1920              // input/output row width (fp32)
#define DSC    512               // scalar channel count (= MUL0)
#define M1     256               // vector channels (MUL1), 3 components
#define M2     128               // tensor channels (MUL2), 5 components
#define NG     384               // gate count (M1 + M2)
#define ROWS   32                // rows per workgroup (2 x 16-row WMMA tiles)
#define LDSW   (ROWS * D_IN)     // staged tile size in halfs
#define NW0    (512 * 896)
#define NW1    (256 * 256)
#define NW2    (128 * 128)

// ---------------------------------------------------------------------------
// WMMA helpers (CDNA5 wave32, V_WMMA_F32_16X16X32_F16)
// ---------------------------------------------------------------------------
__device__ __forceinline__ v8f wmma_f16(v16h a, v16h b, v8f c) {
  // 8 args: (neg_a, A, neg_b, B, c_mod, C, reuse_a, reuse_b)
  return __builtin_amdgcn_wmma_f32_16x16x32_f16(false, a, false, b, (short)0, c,
                                                false, false);
}

__device__ __forceinline__ v16h frag2(const h8* p, int lo, int hi) {
  h8 a = p[lo], b = p[hi];
  return __builtin_shufflevector(a, b, 0, 1, 2, 3, 4, 5, 6, 7,
                                        8, 9, 10, 11, 12, 13, 14, 15);
}

// One 16x16 output tile, K-contiguous f16 A (LDS) and B (global, [O][K]).
// A fragment per lane (kg = lane>>4): K = kg*8+0..7 and 16+kg*8+0..7
// B fragment per lane:                K = kg*8+0..7 and 16+kg*8+0..7
// Software-pipelined: next fragments are loaded before the current WMMA so
// waits stay non-zero and L2 latency hides under the matrix op.
template <int K>
__device__ __forceinline__ v8f gemm_tile(const _Float16* aRow,
                                         const _Float16* bCol, int kg) {
  v8f acc = {};
  const h8* ap = (const h8*)aRow;
  const h8* bp = (const h8*)bCol;
  v16h A = frag2(ap, kg, 2 + kg);
  v16h B = frag2(bp, kg, 2 + kg);
#pragma unroll 4
  for (int k0 = 32; k0 < K; k0 += 32) {
    const int i = k0 >> 3;
    v16h An = frag2(ap, i + kg, i + 2 + kg);
    v16h Bn = frag2(bp, i + kg, i + 2 + kg);
    acc = wmma_f16(A, B, acc);
    A = An;
    B = Bn;
  }
  return wmma_f16(A, B, acc);
}

__device__ __forceinline__ float fast_sigmoid(float s) {
  return __builtin_amdgcn_rcpf(1.0f + __expf(-s));  // v_rcp_f32, no div-scale
}

// ---------------------------------------------------------------------------
// Prepass: convert weights to f16 and transpose to [O][K] so B fragments are
// two contiguous b128 loads. wt: W0t[896][512] | W1t[256][256] | W2t[128][128]
// ---------------------------------------------------------------------------
__global__ void __launch_bounds__(256)
wcvt_kernel(const float* __restrict__ W0, const float* __restrict__ W1,
            const float* __restrict__ W2, _Float16* __restrict__ wt) {
  long i = (long)blockIdx.x * 256 + threadIdx.x;
  if (i < NW0) {
    long o = i / 512, k = i % 512;
    wt[i] = (_Float16)W0[k * 896 + o];
  } else if (i < NW0 + NW1) {
    long j = i - NW0;
    long o = j / 256, k = j % 256;
    wt[NW0 + j] = (_Float16)W1[k * 256 + o];
  } else if (i < NW0 + NW1 + NW2) {
    long j = i - NW0 - NW1;
    long o = j / 128, k = j % 128;
    wt[NW0 + NW1 + j] = (_Float16)W2[k * 128 + o];
  }
}

// ---------------------------------------------------------------------------
// Main fused kernel: stage rows -> GEMM0 (+gates) -> barrier -> gated GEMM1/2
// ---------------------------------------------------------------------------
__global__ void __launch_bounds__(256)
eqmlp_main(const float* __restrict__ x, const _Float16* __restrict__ wt,
           float* __restrict__ out, int N) {
  extern __shared__ _Float16 smem[];
  _Float16* xs = smem;              // ROWS * D_IN halfs, v/t de-interleaved
  _Float16* gs = smem + LDSW;       // ROWS * NG half gates

  const _Float16* W0t = wt;
  const _Float16* W1t = wt + NW0;
  const _Float16* W2t = wt + NW0 + NW1;

  const int tid  = threadIdx.x;
  const int wave = tid >> 5;
  const int lane = tid & 31;
  const int lr   = lane & 15;
  const int kg   = lane >> 4;
  const long rowBase = (long)blockIdx.x * ROWS;
  if (rowBase >= N) return;

  // ---- Phase A: stage 32x1920 fp32 tile as de-interleaved f16 ----
  // dst col: s -> c; v (i*3+m) -> 512 + m*256 + i; t (i*5+m) -> 1280 + m*128 + i
  // Vectorized: each thread streams float4 (coalesced), scatters 4 f16 to LDS.
  {
    int r = 0, c4 = tid * 4;  // D_IN % 4 == 0 so float4 never crosses a row
    for (int e = tid; e < (ROWS * D_IN) / 4; e += 256) {
      float4 f = make_float4(0.f, 0.f, 0.f, 0.f);
      if (rowBase + r < N)
        f = *(const float4*)(x + (rowBase + r) * (long)D_IN + c4);
      const float fv[4] = {f.x, f.y, f.z, f.w};
#pragma unroll
      for (int j = 0; j < 4; ++j) {
        const int c = c4 + j;
        int d;
        if (c < DSC) {
          d = c;
        } else if (c < DSC + 3 * M1) {
          int u = c - DSC;
          d = DSC + (u % 3) * M1 + u / 3;
        } else {
          int u = c - DSC - 3 * M1;
          d = DSC + 3 * M1 + (u % 5) * M2 + u / 5;
        }
        xs[r * D_IN + d] = (_Float16)fv[j];
      }
      c4 += 1024;
      if (c4 >= D_IN) { c4 -= D_IN; ++r; }
    }
  }
  __syncthreads();

  // ---- Phase B: s @ W0 (K=512), 2 row-tiles x 56 col-tiles, round robin ----
  for (int tt = wave; tt < 2 * 56; tt += 8) {
    const int rt = tt / 56, ct = tt % 56;
    const int colBase = ct * 16, col = colBase + lr;
    const _Float16* aRow = xs + (rt * 16 + lr) * D_IN;
    const _Float16* bCol = W0t + (size_t)col * DSC;
    // warm L2/WGP$ for this wave's next tile while we compute
    if (tt + 8 < 2 * 56) {
      const int ncol = ((tt + 8) % 56) * 16 + lr;
      __builtin_prefetch(W0t + (size_t)ncol * DSC, 0, 1);
      __builtin_prefetch(W0t + (size_t)ncol * DSC + 256, 0, 1);
    }
    v8f acc = gemm_tile<DSC>(aRow, bCol, kg);
    if (colBase < DSC) {  // scalar outputs: leaky_relu
#pragma unroll
      for (int r8 = 0; r8 < 8; ++r8) {
        const int mrow = rt * 16 + r8 + 8 * kg;   // C layout: M = r + 8*kg
        const long grow = rowBase + mrow;
        float v = acc[r8] * 0.04419417382415922f;  // 1/sqrt(512)
        v = (v > 0.0f) ? v : 0.01f * v;
        if (grow < N) out[grow * D_IN + col] = v;
      }
    } else {  // gate outputs: sigmoid -> LDS
      const int gc = col - DSC;
#pragma unroll
      for (int r8 = 0; r8 < 8; ++r8) {
        const int mrow = rt * 16 + r8 + 8 * kg;
        float s = acc[r8] * 0.04419417382415922f;
        gs[mrow * NG + gc] = (_Float16)fast_sigmoid(s);
      }
    }
  }
  __syncthreads();

  // ---- Phase C: gated v @ W1 (48 tiles/rt) and t @ W2 (40 tiles/rt) ----
  for (int tt = wave; tt < 2 * 88; tt += 8) {
    const int rt = tt / 88, u = tt % 88;
    if (u < 48) {  // v path: K=256, component m = u>>4
      const int m = u >> 4, ct = u & 15;
      const int col = ct * 16 + lr;
      const _Float16* aRow = xs + (rt * 16 + lr) * D_IN + DSC + m * M1;
      const _Float16* bCol = W1t + (size_t)col * M1;
      v8f acc = gemm_tile<M1>(aRow, bCol, kg);
#pragma unroll
      for (int r8 = 0; r8 < 8; ++r8) {
        const int mrow = rt * 16 + r8 + 8 * kg;
        const long grow = rowBase + mrow;
        float g = (float)gs[mrow * NG + col];
        if (grow < N)
          out[grow * D_IN + DSC + col * 3 + m] = acc[r8] * 0.0625f * g;
      }
    } else {  // t path: K=128, component m = (u-48)>>3
      const int u2 = u - 48;
      const int m = u2 >> 3, ct = u2 & 7;
      const int col = ct * 16 + lr;
      const _Float16* aRow = xs + (rt * 16 + lr) * D_IN + DSC + 3 * M1 + m * M2;
      const _Float16* bCol = W2t + (size_t)col * M2;
      v8f acc = gemm_tile<M2>(aRow, bCol, kg);
#pragma unroll
      for (int r8 = 0; r8 < 8; ++r8) {
        const int mrow = rt * 16 + r8 + 8 * kg;
        const long grow = rowBase + mrow;
        float g = (float)gs[mrow * NG + M1 + col];
        if (grow < N)
          out[grow * D_IN + DSC + 3 * M1 + col * 5 + m] =
              acc[r8] * 0.08838834764831845f * g;  // 1/sqrt(128)
      }
    }
  }
}

// ---------------------------------------------------------------------------
extern "C" void kernel_launch(void* const* d_in, const int* in_sizes, int n_in,
                              void* d_out, int out_size, void* d_ws,
                              size_t ws_size, hipStream_t stream) {
  const float* x  = (const float*)d_in[0];
  const float* W0 = (const float*)d_in[1];
  const float* W1 = (const float*)d_in[2];
  const float* W2 = (const float*)d_in[3];
  float* out = (float*)d_out;
  _Float16* wt = (_Float16*)d_ws;  // needs (NW0+NW1+NW2)*2 ~= 1.03 MB scratch

  const int N = in_sizes[0] / D_IN;

  // Prepass: f16 + transpose weights into scratch (stream-ordered).
  const int cvtTotal = NW0 + NW1 + NW2;
  wcvt_kernel<<<dim3((cvtTotal + 255) / 256), dim3(256), 0, stream>>>(W0, W1,
                                                                      W2, wt);

  const size_t shmem = (size_t)(LDSW + ROWS * NG) * sizeof(_Float16);  // 144 KB
  hipFuncSetAttribute((const void*)eqmlp_main,
                      hipFuncAttributeMaxDynamicSharedMemorySize, (int)shmem);
  eqmlp_main<<<dim3((N + ROWS - 1) / ROWS), dim3(256), shmem, stream>>>(x, wt,
                                                                       out, N);
}
